// EGATGCNConv_41721312313495
// MI455X (gfx1250) — compile-verified, matrix operand order checked by reading, not run on hardware
//
#include <hip/hip_runtime.h>
#include <stdint.h>
#include <stddef.h>

// Problem constants (from reference)
#define GN   10000
#define GE   160000
#define INN  256
#define INE  128
#define CC   64
#define HH   4
#define HC   256

typedef __attribute__((ext_vector_type(16))) __bf16 v16bf;
typedef __attribute__((ext_vector_type(8)))  float  v8f;
typedef __attribute__((__vector_size__(4 * sizeof(int)))) int vi4;

union Frag { v16bf v; uint32_t u[8]; };

#define BM   128
#define BN   64
#define BK   32
#define LDF  36   // f32 row stride for K=32 tiles (pad 4 -> 144B rows, 16B aligned)

#define AS1 __attribute__((address_space(1)))
#define AS3 __attribute__((address_space(3)))

#if defined(__AMDGCN__) && __has_builtin(__builtin_amdgcn_global_load_async_to_lds_b128) && __has_builtin(__builtin_amdgcn_global_load_async_to_lds_b32)
#define HAVE_ASYNC_LDS 1
#else
#define HAVE_ASYNC_LDS 0
#endif

__device__ __forceinline__ v8f wmma_bf16(const Frag& a, const Frag& b, v8f c) {
  return __builtin_amdgcn_wmma_f32_16x16x32_bf16(false, a.v, false, b.v,
                                                 (short)0, c, false, false);
}

// pack two f32 (K, K+1) into one dword of bf16 pair (low = K) -> v_cvt_pk_bf16_f32
__device__ __forceinline__ uint32_t pkbf(float x, float y) {
  union { __bf16 h[2]; uint32_t u; } r;
  r.h[0] = (__bf16)x; r.h[1] = (__bf16)y;
  return r.u;
}

// Stage one BK-step of A (contiguous) and B (transposed) fp32 tiles into LDS.
// Async global->LDS on gfx1250 (ASYNCcnt) with a portable fallback.
__device__ __forceinline__ void stage_tiles(const float* __restrict__ arow,
                                            const float* __restrict__ B,
                                            float* Asf, float* Bsf,
                                            int k0, int K, int Nd, int col0,
                                            int sr, int sk, int bn, int bk) {
#if HAVE_ASYNC_LDS
  {
    const float* g = arow + k0 + sk;      // 16B-aligned (K mult of 4, sk mult of 16)
    float* l = &Asf[sr * LDF + sk];
#pragma unroll
    for (int i = 0; i < 4; ++i)
      __builtin_amdgcn_global_load_async_to_lds_b128(
          (AS1 vi4*)(g + i * 4), (AS3 vi4*)(l + i * 4), 0, 0);
  }
#pragma unroll
  for (int i = 0; i < 8; ++i)
    __builtin_amdgcn_global_load_async_to_lds_b32(
        (AS1 int*)(B + (size_t)(k0 + bk + i) * Nd + col0 + bn),
        (AS3 int*)&Bsf[bn * LDF + bk + i], 0, 0);
  asm volatile("s_wait_asynccnt 0" ::: "memory");
#else
  {
    const float* g = arow + k0 + sk;
    float4* l = (float4*)&Asf[sr * LDF + sk];
#pragma unroll
    for (int i = 0; i < 4; ++i) l[i] = ((const float4*)g)[i];
  }
#pragma unroll
  for (int i = 0; i < 8; ++i)
    Bsf[bn * LDF + bk + i] = B[(size_t)(k0 + bk + i) * Nd + col0 + bn];
#endif
  if (k0 + BK < K) __builtin_prefetch(arow + k0 + BK + sk, 0, 1);
}

// ---------------------------------------------------------------------------
// Tiled GEMM: C[M,Nd] = act(A[M,K] @ B[K,Nd] + bias) + resid
// fp32 in HBM -> fp32 LDS tiles (async DMA) -> bf16 fragments -> f32 WMMA.
// 256 threads = 8 waves; 128x64 block tile; each wave: 4 M-tiles x 1 N-tile
// sharing one B fragment => 4 v_wmma per K-step per wave.
// ---------------------------------------------------------------------------
__global__ __launch_bounds__(256)
void gemm_bf16_wmma(const float* __restrict__ A, const float* __restrict__ B,
                    const float* __restrict__ bias, const float* __restrict__ resid,
                    float* __restrict__ Cmat, int M, int Nd, int K, int act)
{
  __shared__ float Asf[BM * LDF];
  __shared__ float Bsf[BN * LDF];
  const int tid  = threadIdx.x;
  const int lane = tid & 31;
  const int wave = tid >> 5;
  const int wcol = wave & 3;    // N tile 0..3
  const int wrow = wave >> 2;   // 64-row half 0..1
  const int hlf  = lane >> 4;
  const int l16  = lane & 15;
  const int row0 = blockIdx.x * BM;
  const int col0 = blockIdx.y * BN;

  const int sr = tid >> 1;          // A staging: row 0..127
  const int sk = (tid & 1) * 16;    // A staging: k chunk
  const int bn = tid >> 2;          // B staging: column 0..63
  const int bk = (tid & 3) * 8;     // B staging: k chunk

  int gar = row0 + sr; if (gar >= M) gar = M - 1;
  const float* arow = A + (size_t)gar * K;

  v8f acc[4] = {};

  for (int k0 = 0; k0 < K; k0 += BK) {
    stage_tiles(arow, B, Asf, Bsf, k0, K, Nd, col0, sr, sk, bn, bk);
    __syncthreads();

    Frag bf;
#pragma unroll
    for (int v = 0; v < 8; ++v) {
      // 16-bit fragment layout (ISA 7.12.2): VGPR v holds K pair at
      // kp = (v>>2)*16 + half*8 + (v&3)*2 ; B mirrored over N.
      const int kp = ((v >> 2) << 4) + (hlf << 3) + ((v & 3) << 1);
      const float* p = &Bsf[(wcol * 16 + l16) * LDF + kp];
      bf.u[v] = pkbf(p[0], p[1]);
    }
#pragma unroll
    for (int j = 0; j < 4; ++j) {
      Frag af;
#pragma unroll
      for (int v = 0; v < 8; ++v) {
        const int kp = ((v >> 2) << 4) + (hlf << 3) + ((v & 3) << 1);
        const float* p = &Asf[(wrow * 64 + j * 16 + l16) * LDF + kp];
        af.u[v] = pkbf(p[0], p[1]);
      }
      acc[j] = wmma_bf16(af, bf, acc[j]);
    }
    __syncthreads();
  }

  const int c = col0 + wcol * 16 + l16;
  const float bval = bias ? bias[c] : 0.0f;
#pragma unroll
  for (int j = 0; j < 4; ++j) {
#pragma unroll
    for (int v = 0; v < 8; ++v) {
      int m = row0 + wrow * 64 + j * 16 + hlf * 8 + v;  // D: M = v + 8*half
      if (m < M) {
        float x = acc[j][v] + bval;
        if (act == 1)      x = fmaxf(x, 0.f);
        else if (act == 2) x = x >= 0.f ? x : 0.2f * x;
        if (resid) x += resid[(size_t)m * Nd + c];
        Cmat[(size_t)m * Nd + c] = x;
      }
    }
  }
}

// ---------------------------------------------------------------------------
// Fused EGAT first layer: C = relu(concat(XNj[dst], Eproj[e], XNi[src]) @ A0_W + b)
// K = 768 in three gathered 256-wide segments (never materializes [E,768]).
// ---------------------------------------------------------------------------
__global__ __launch_bounds__(256)
void egat_a0_wmma(const float* __restrict__ XNi, const float* __restrict__ XNj,
                  const float* __restrict__ Eproj,
                  const int* __restrict__ srcI, const int* __restrict__ dstI,
                  const float* __restrict__ B, const float* __restrict__ bias,
                  float* __restrict__ Cmat, int M, int Nd)
{
  __shared__ float Asf[BM * LDF];
  __shared__ float Bsf[BN * LDF];
  __shared__ int sSrc[BM], sDst[BM], sE[BM];

  const int tid  = threadIdx.x;
  const int lane = tid & 31;
  const int wave = tid >> 5;
  const int wcol = wave & 3;
  const int wrow = wave >> 2;
  const int hlf  = lane >> 4;
  const int l16  = lane & 15;
  const int row0 = blockIdx.x * BM;
  const int col0 = blockIdx.y * BN;
  const int sr = tid >> 1;
  const int sk = (tid & 1) * 16;
  const int bn = tid >> 2;
  const int bk = (tid & 3) * 8;

  if (tid < BM) {
    int e = row0 + tid; if (e >= M) e = M - 1;
    sE[tid] = e; sSrc[tid] = srcI[e]; sDst[tid] = dstI[e];
  }
  __syncthreads();

  v8f acc[4] = {};

  for (int k0 = 0; k0 < 3 * HC; k0 += BK) {
    const int seg  = k0 >> 8;
    const int kloc = (k0 & 255) + sk;
    const float* base;
    if (seg == 0)      base = XNj   + (size_t)sDst[sr] * HC;  // x_i
    else if (seg == 1) base = Eproj + (size_t)sE[sr]   * HC;  // e_ij
    else               base = XNi   + (size_t)sSrc[sr] * HC;  // x_j
#if HAVE_ASYNC_LDS
    {
      const float* g = base + kloc;
      float* l = &Asf[sr * LDF + sk];
#pragma unroll
      for (int i = 0; i < 4; ++i)
        __builtin_amdgcn_global_load_async_to_lds_b128(
            (AS1 vi4*)(g + i * 4), (AS3 vi4*)(l + i * 4), 0, 0);
    }
#pragma unroll
    for (int i = 0; i < 8; ++i)
      __builtin_amdgcn_global_load_async_to_lds_b32(
          (AS1 int*)(B + (size_t)(k0 + bk + i) * Nd + col0 + bn),
          (AS3 int*)&Bsf[bn * LDF + bk + i], 0, 0);
    asm volatile("s_wait_asynccnt 0" ::: "memory");
#else
    {
      const float* g = base + kloc;
      float4* l = (float4*)&Asf[sr * LDF + sk];
#pragma unroll
      for (int i = 0; i < 4; ++i) l[i] = ((const float4*)g)[i];
    }
#pragma unroll
    for (int i = 0; i < 8; ++i)
      Bsf[bn * LDF + bk + i] = B[(size_t)(k0 + bk + i) * Nd + col0 + bn];
#endif
    __syncthreads();

    Frag bf;
#pragma unroll
    for (int v = 0; v < 8; ++v) {
      const int kp = ((v >> 2) << 4) + (hlf << 3) + ((v & 3) << 1);
      const float* p = &Bsf[(wcol * 16 + l16) * LDF + kp];
      bf.u[v] = pkbf(p[0], p[1]);
    }
#pragma unroll
    for (int j = 0; j < 4; ++j) {
      Frag af;
#pragma unroll
      for (int v = 0; v < 8; ++v) {
        const int kp = ((v >> 2) << 4) + (hlf << 3) + ((v & 3) << 1);
        const float* p = &Asf[(wrow * 64 + j * 16 + l16) * LDF + kp];
        af.u[v] = pkbf(p[0], p[1]);
      }
      acc[j] = wmma_bf16(af, bf, acc[j]);
    }
    __syncthreads();
  }

  const int c = col0 + wcol * 16 + l16;
  const float bval = bias[c];
#pragma unroll
  for (int j = 0; j < 4; ++j) {
#pragma unroll
    for (int v = 0; v < 8; ++v) {
      int m = row0 + wrow * 64 + j * 16 + hlf * 8 + v;
      if (m < M) Cmat[(size_t)m * Nd + c] = fmaxf(acc[j][v] + bval, 0.f);
    }
  }
}

// ---------------------------------------------------------------------------
// Elementwise / scatter kernels
// ---------------------------------------------------------------------------
__global__ void fill_f32(float* p, float v, int n) {
  int i = blockIdx.x * 256 + threadIdx.x; if (i < n) p[i] = v;
}
__global__ void fill_u32(unsigned* p, unsigned v, int n) {
  int i = blockIdx.x * 256 + threadIdx.x; if (i < n) p[i] = v;
}

// monotone float<->uint encoding for atomic max over signed floats
__device__ __forceinline__ unsigned encf(float x) {
  unsigned b = __float_as_uint(x);
  return (b & 0x80000000u) ? ~b : (b | 0x80000000u);
}
__device__ __forceinline__ float decf(unsigned k) {
  unsigned b = (k & 0x80000000u) ? (k ^ 0x80000000u) : ~k;
  return __uint_as_float(b);
}

__global__ void scores_k(const float* __restrict__ f, const float* __restrict__ ap,
                         float* __restrict__ sc) {
  int i = blockIdx.x * 256 + threadIdx.x;
  if (i >= GE * HH) return;
  int e = i >> 2, h = i & 3;
  const float* fp = f + (size_t)e * HC + h * CC;
  const float* a  = ap + h * CC;
  float s = 0.f;
#pragma unroll
  for (int c2 = 0; c2 < CC; ++c2) s += fp[c2] * a[c2];
  sc[i] = s;
}

__global__ void smax_k(const float* __restrict__ sc, const int* __restrict__ dstI,
                       unsigned* __restrict__ smax) {
  int i = blockIdx.x * 256 + threadIdx.x;
  if (i >= GE * HH) return;
  int e = i >> 2, h = i & 3;
  atomicMax(&smax[dstI[e] * HH + h], encf(sc[i]));
}

__global__ void exden_k(const float* __restrict__ sc, const int* __restrict__ dstI,
                        const unsigned* __restrict__ smax,
                        float* __restrict__ ex, float* __restrict__ den) {
  int i = blockIdx.x * 256 + threadIdx.x;
  if (i >= GE * HH) return;
  int e = i >> 2, h = i & 3;
  int d = dstI[e];
  float v = __expf(sc[i] - decf(smax[d * HH + h]));
  ex[i] = v;
  atomicAdd(&den[d * HH + h], v);
}

__global__ void alpha_k(const float* __restrict__ ex, const float* __restrict__ den,
                        const int* __restrict__ dstI, const float* __restrict__ linW,
                        float* __restrict__ alpha, float* __restrict__ attnOut) {
  int e = blockIdx.x * 256 + threadIdx.x;
  if (e >= GE) return;
  int d = dstI[e];
  float aw = 0.f;
#pragma unroll
  for (int h = 0; h < HH; ++h) {
    float a = ex[e * HH + h] / den[d * HH + h];
    alpha[e * HH + h] = a;
    aw += a * linW[h];
  }
  attnOut[e] = aw;
}

__global__ void msg_k(const float* __restrict__ XNi, const int* __restrict__ srcI,
                      const int* __restrict__ dstI, const float* __restrict__ alpha,
                      float* __restrict__ agg) {
  long long i = (long long)blockIdx.x * 256 + threadIdx.x;
  if (i >= (long long)GE * HC) return;
  int e  = (int)(i >> 8);
  int hc = (int)(i & 255);
  int h  = hc >> 6;
  float m = XNi[(size_t)srcI[e] * HC + hc] * alpha[e * HH + h];
  atomicAdd(&agg[(size_t)dstI[e] * HC + hc], m);
}

__global__ void deg_k(const int* __restrict__ srcI, const int* __restrict__ dstI,
                      float* __restrict__ deg) {
  int e = blockIdx.x * 256 + threadIdx.x;
  if (e >= GE) return;
  if (srcI[e] != dstI[e]) atomicAdd(&deg[srcI[e]], 1.0f);
}
__global__ void dinv_k(float* __restrict__ deg) {
  int i = blockIdx.x * 256 + threadIdx.x;
  if (i >= GN) return;
  deg[i] = rsqrtf(deg[i] + 1.0f);   // +1 for the added self loop; deg >= 1
}

__global__ void gcn_edge_k(const float* __restrict__ xg, const int* __restrict__ srcI,
                           const int* __restrict__ dstI, const float* __restrict__ dinv,
                           float* __restrict__ outN) {
  long long i = (long long)blockIdx.x * 256 + threadIdx.x;
  if (i >= (long long)GE * CC) return;
  int e = (int)(i >> 6), c = (int)(i & 63);
  int s = srcI[e], d = dstI[e];
  if (s == d) return;  // removed self loops carry weight 0
  atomicAdd(&outN[(size_t)d * CC + c], dinv[s] * dinv[d] * xg[(size_t)s * CC + c]);
}
__global__ void gcn_self_k(const float* __restrict__ xg, const float* __restrict__ dinv,
                           const float* __restrict__ gb, float* __restrict__ outN) {
  int i = blockIdx.x * 256 + threadIdx.x;
  if (i >= GN * CC) return;
  int n = i >> 6, c = i & 63;
  outN[i] += dinv[n] * dinv[n] * xg[i] + gb[c];
}

// ---------------------------------------------------------------------------
extern "C" void kernel_launch(void* const* d_in, const int* in_sizes, int n_in,
                              void* d_out, int out_size, void* d_ws, size_t ws_size,
                              hipStream_t stream)
{
  (void)in_sizes; (void)n_in; (void)out_size; (void)ws_size;

  const float* nf    = (const float*)d_in[0];
  const float* ef    = (const float*)d_in[1];
  const int*   eidx  = (const int*)d_in[2];
  const float* W_ni  = (const float*)d_in[3];
  const float* W_nj  = (const float*)d_in[4];
  const float* W_eij = (const float*)d_in[5];
  const float* W_nt  = (const float*)d_in[6];
  const float* b_nt  = (const float*)d_in[7];
  const float* W_et  = (const float*)d_in[8];
  const float* b_et  = (const float*)d_in[9];
  const float* A0W   = (const float*)d_in[10];
  const float* A0b   = (const float*)d_in[11];
  const float* A1W   = (const float*)d_in[12];
  const float* A1b   = (const float*)d_in[13];
  const float* A2W   = (const float*)d_in[14];
  const float* A2b   = (const float*)d_in[15];
  const float* attnP = (const float*)d_in[16];
  const float* Nm0W  = (const float*)d_in[17];
  const float* Nm0b  = (const float*)d_in[18];
  const float* Nm1W  = (const float*)d_in[19];
  const float* Nm1b  = (const float*)d_in[20];
  const float* Nm2W  = (const float*)d_in[21];
  const float* Nm2b  = (const float*)d_in[22];
  const float* Em0W  = (const float*)d_in[23];
  const float* Em0b  = (const float*)d_in[24];
  const float* Em1W  = (const float*)d_in[25];
  const float* Em1b  = (const float*)d_in[26];
  const float* Em2W  = (const float*)d_in[27];
  const float* Em2b  = (const float*)d_in[28];
  const float* linW  = (const float*)d_in[29];
  const float* gcnW  = (const float*)d_in[30];
  const float* gcnb  = (const float*)d_in[31];

  const int* srcI = eidx;
  const int* dstI = eidx + GE;

  float* outN = (float*)d_out;                  // [GN, CC]
  float* outE = outN + (size_t)GN * CC;         // [GE, CC]
  float* outA = outE + (size_t)GE * CC;         // [GE]

  float* ws = (float*)d_ws;
  size_t off = 0;
  auto take = [&](size_t n) { float* p = ws + off; off += n; return p; };
  float* XNi   = take((size_t)GN * HC);   // node @ W_ni  (x_j source)
  float* XNj   = take((size_t)GN * HC);   // node @ W_nj  (x_i source)
  float* nodeT = take((size_t)GN * CC);
  float* Eproj = take((size_t)GE * HC);
  float* edgeT = take((size_t)GE * CC);
  float* buf1  = take((size_t)GE * HC);
  float* buf2  = take((size_t)GE * HC);
  float* ebufA = take((size_t)GE * CC);
  float* ebufB = take((size_t)GE * CC);
  float* sc    = take((size_t)GE * HH);
  float* exb   = take((size_t)GE * HH);
  float* alphb = take((size_t)GE * HH);
  unsigned* smax = (unsigned*)take((size_t)GN * HH);
  float* den   = take((size_t)GN * HH);
  float* agg   = take((size_t)GN * HC);
  float* nbA   = take((size_t)GN * CC);
  float* nbB   = take((size_t)GN * CC);
  float* nodeO = take((size_t)GN * CC);
  float* xg    = take((size_t)GN * CC);
  float* deg   = take((size_t)GN);

  auto cdiv = [](int a, int b) { return (a + b - 1) / b; };
  const int NB_N = cdiv(GN, BM), NB_E = cdiv(GE, BM);

  // zero / init accumulators (d_out & d_ws are poisoned by the harness)
  fill_f32<<<cdiv(GN * CC, 256), 256, 0, stream>>>(outN, 0.f, GN * CC);
  fill_f32<<<cdiv(GN * HC, 256), 256, 0, stream>>>(agg, 0.f, GN * HC);
  fill_f32<<<cdiv(GN * HH, 256), 256, 0, stream>>>(den, 0.f, GN * HH);
  fill_f32<<<cdiv(GN, 256), 256, 0, stream>>>(deg, 0.f, GN);
  fill_u32<<<cdiv(GN * HH, 256), 256, 0, stream>>>(smax, 0x007FFFFFu, GN * HH); // enc(-inf)

  // input projections (WMMA)
  gemm_bf16_wmma<<<dim3(NB_N, HC / BN), 256, 0, stream>>>(nf, W_ni, nullptr, nullptr, XNi, GN, HC, INN, 0);
  gemm_bf16_wmma<<<dim3(NB_N, HC / BN), 256, 0, stream>>>(nf, W_nj, nullptr, nullptr, XNj, GN, HC, INN, 0);
  gemm_bf16_wmma<<<dim3(NB_N, CC / BN), 256, 0, stream>>>(nf, W_nt, b_nt, nullptr, nodeT, GN, CC, INN, 0);
  gemm_bf16_wmma<<<dim3(NB_E, HC / BN), 256, 0, stream>>>(ef, W_eij, nullptr, nullptr, Eproj, GE, HC, INE, 0);
  gemm_bf16_wmma<<<dim3(NB_E, CC / BN), 256, 0, stream>>>(ef, W_et, b_et, nullptr, edgeT, GE, CC, INE, 0);

  // EGAT attention MLP (gather-fused first layer; relu each layer; the
  // trailing leaky_relu is identity on relu output)
  egat_a0_wmma  <<<dim3(NB_E, HC / BN), 256, 0, stream>>>(XNi, XNj, Eproj, srcI, dstI, A0W, A0b, buf1, GE, HC);
  gemm_bf16_wmma<<<dim3(NB_E, HC / BN), 256, 0, stream>>>(buf1, A1W, A1b, nullptr, buf2, GE, HC, HC, 1);
  gemm_bf16_wmma<<<dim3(NB_E, HC / BN), 256, 0, stream>>>(buf2, A2W, A2b, nullptr, buf1, GE, HC, HC, 1); // f

  // attention scores + segment softmax over dst
  scores_k<<<cdiv(GE * HH, 256), 256, 0, stream>>>(buf1, attnP, sc);
  smax_k  <<<cdiv(GE * HH, 256), 256, 0, stream>>>(sc, dstI, smax);
  exden_k <<<cdiv(GE * HH, 256), 256, 0, stream>>>(sc, dstI, smax, exb, den);
  alpha_k <<<cdiv(GE, 256), 256, 0, stream>>>(exb, den, dstI, linW, alphb, outA);

  // message aggregation: agg[dst] += x_j * alpha
  msg_k<<<(GE * HC) / 256, 256, 0, stream>>>(XNi, srcI, dstI, alphb, agg);

  // node MLP + residual
  gemm_bf16_wmma<<<dim3(NB_N, CC / BN), 256, 0, stream>>>(agg, Nm0W, Nm0b, nullptr, nbA, GN, CC, HC, 1);
  gemm_bf16_wmma<<<dim3(NB_N, CC / BN), 256, 0, stream>>>(nbA, Nm1W, Nm1b, nullptr, nbB, GN, CC, CC, 1);
  gemm_bf16_wmma<<<dim3(NB_N, CC / BN), 256, 0, stream>>>(nbB, Nm2W, Nm2b, nodeT, nodeO, GN, CC, CC, 1);

  // edge MLP + residual -> edge_out section of d_out
  gemm_bf16_wmma<<<dim3(NB_E, CC / BN), 256, 0, stream>>>(buf1, Em0W, Em0b, nullptr, ebufA, GE, CC, HC, 1);
  gemm_bf16_wmma<<<dim3(NB_E, CC / BN), 256, 0, stream>>>(ebufA, Em1W, Em1b, nullptr, ebufB, GE, CC, CC, 1);
  gemm_bf16_wmma<<<dim3(NB_E, CC / BN), 256, 0, stream>>>(ebufB, Em2W, Em2b, edgeT, outE, GE, CC, CC, 1);

  // GCN: xg = node_out @ gcn_W, symmetric-normalized aggregation + bias
  gemm_bf16_wmma<<<dim3(NB_N, CC / BN), 256, 0, stream>>>(nodeO, gcnW, nullptr, nullptr, xg, GN, CC, CC, 0);
  deg_k <<<cdiv(GE, 256), 256, 0, stream>>>(srcI, dstI, deg);
  dinv_k<<<cdiv(GN, 256), 256, 0, stream>>>(deg);
  gcn_edge_k<<<(GE * CC) / 256, 256, 0, stream>>>(xg, srcI, dstI, deg, outN);
  gcn_self_k<<<cdiv(GN * CC, 256), 256, 0, stream>>>(xg, deg, gcnb, outN);
}
